// PyramidRetNet_9783935500542
// MI455X (gfx1250) — compile-verified, matrix operand order checked by reading, not run on hardware
//
#include <hip/hip_runtime.h>
#include <math.h>

typedef float  v8f    __attribute__((ext_vector_type(8)));
typedef __bf16 v16bf  __attribute__((ext_vector_type(16)));
typedef __bf16 bf16x2 __attribute__((ext_vector_type(2)));
typedef int    v4i    __attribute__((__vector_size__(16)));

#define GLOBAL_AS __attribute__((address_space(1)))
#define LDS_AS    __attribute__((address_space(3)))

#if __has_builtin(__builtin_amdgcn_global_load_async_to_lds_b128)
#define HAVE_ASYNC_LDS 1
#else
#define HAVE_ASYNC_LDS 0
#endif

// ---------------------------------------------------------------------------
// helpers
// ---------------------------------------------------------------------------
__device__ inline float wave_sum(float v) {
#pragma unroll
  for (int m = 16; m; m >>= 1) v += __shfl_xor(v, m, 32);
  return v;
}

__device__ inline v8f v8f_zero() {
  v8f z = {0.f, 0.f, 0.f, 0.f, 0.f, 0.f, 0.f, 0.f};
  return z;
}

// K-pair base offset inside a 16-bit A/B WMMA fragment (CDNA5 layout):
// VGPR j holds K pair {base, base+1}; lanes 16..31 shift K by +8.
__device__ inline int kpair_base(int j, int kh) {
  return ((j < 4) ? (2 * j) : (16 + 2 * (j - 4))) + kh;
}

__device__ inline v16bf load_frag_bf(const __bf16* __restrict__ p, int kh) {
  v16bf f;
#pragma unroll
  for (int j = 0; j < 8; ++j) {
    int k = kpair_base(j, kh);
    f[2 * j]     = p[k];
    f[2 * j + 1] = p[k + 1];
  }
  return f;
}

__device__ inline v8f wmma_bf16(v16bf a, v16bf b, v8f c) {
  return __builtin_amdgcn_wmma_f32_16x16x32_bf16(false, a, false, b, (short)0, c,
                                                 false, false);
}

__device__ inline float silu_f(float x) { return x / (1.f + expf(-x)); }
__device__ inline float gelu_f(float x) {
  return 0.5f * x * (1.f + erff(x * 0.70710678118654752f));
}

// ---------------------------------------------------------------------------
// generic bf16-WMMA GEMM: out[M,Ncol] = epi(A[M,K] @ Wt[K,Ncol] + bias)
// EPI: 0=bias 1=bias*scale 2=silu(bias) 3=gelu(bias) 4=bias + res
// ---------------------------------------------------------------------------
template <int EPI>
__global__ void __launch_bounds__(256) gemm_kernel(
    const float* __restrict__ A, const float* __restrict__ Wt,
    const float* __restrict__ bias, const float* __restrict__ res,
    float* __restrict__ out, int M, int Ncol, int K, float postScale) {
  __shared__ __bf16 As[128][40];
  __shared__ __bf16 Bs[64][40];

  const int tid  = threadIdx.x;
  const int n0   = blockIdx.x * 64;
  const int m0   = blockIdx.y * 128;
  const int wave = tid >> 5;
  const int lane = tid & 31;
  const int l16  = lane & 15;
  const int kh   = (lane & 16) ? 8 : 0;
  const int wy   = wave >> 1;   // 0..3 -> 32-row strip
  const int wx   = wave & 1;    // 0..1 -> 32-col strip

  v8f acc00 = v8f_zero(), acc01 = v8f_zero(), acc10 = v8f_zero(), acc11 = v8f_zero();

  for (int k0 = 0; k0 < K; k0 += 32) {
    if (k0 + 32 < K) __builtin_prefetch(&A[(size_t)m0 * K + k0 + 32], 0, 1);
    // A tile: 128x32 f32 -> bf16, packed pair stores (2048 pairs / 256 thr)
#pragma unroll
    for (int i = 0; i < 8; ++i) {
      int e = tid + i * 256;
      int r = e >> 4, c2 = e & 15;
      const float2 v2 =
          *(const float2*)(A + (size_t)(m0 + r) * K + k0 + c2 * 2);
      bf16x2 p;
      p[0] = (__bf16)v2.x;
      p[1] = (__bf16)v2.y;
      *(bf16x2*)(&As[r][c2 * 2]) = p;
    }
    // B tile: 32x64 f32 -> bf16 transposed; pack along K (1024 pairs)
#pragma unroll
    for (int i = 0; i < 4; ++i) {
      int e = tid + i * 256;
      int c = e & 63, rp = e >> 6;  // rp: 0..15 -> k pair
      float w0 = Wt[(size_t)(k0 + 2 * rp) * Ncol + n0 + c];
      float w1 = Wt[(size_t)(k0 + 2 * rp + 1) * Ncol + n0 + c];
      bf16x2 p;
      p[0] = (__bf16)w0;
      p[1] = (__bf16)w1;
      *(bf16x2*)(&Bs[c][2 * rp]) = p;
    }
    __syncthreads();

    v16bf a0 = load_frag_bf(&As[wy * 32 + l16][0], kh);
    v16bf a1 = load_frag_bf(&As[wy * 32 + 16 + l16][0], kh);
    v16bf b0 = load_frag_bf(&Bs[wx * 32 + l16][0], kh);
    v16bf b1 = load_frag_bf(&Bs[wx * 32 + 16 + l16][0], kh);

    acc00 = wmma_bf16(a0, b0, acc00);
    acc01 = wmma_bf16(a0, b1, acc01);
    acc10 = wmma_bf16(a1, b0, acc10);
    acc11 = wmma_bf16(a1, b1, acc11);
    __syncthreads();
  }

  auto store_tile = [&](v8f acc, int mi, int ni) {
#pragma unroll
    for (int r = 0; r < 8; ++r) {
      int gm = m0 + wy * 32 + mi * 16 + r + ((lane & 16) ? 8 : 0);
      int gn = n0 + wx * 32 + ni * 16 + l16;
      float v = acc[r] + bias[gn];
      if (EPI == 1) v *= postScale;
      if (EPI == 2) v = silu_f(v);
      if (EPI == 3) v = gelu_f(v);
      if (EPI == 4) v += res[(size_t)gm * Ncol + gn];
      out[(size_t)gm * Ncol + gn] = v;
    }
  };
  store_tile(acc00, 0, 0);
  store_tile(acc01, 0, 1);
  store_tile(acc10, 1, 0);
  store_tile(acc11, 1, 1);
}

// ---------------------------------------------------------------------------
// fused retention:  o[h] = ((qr_h @ kr_h^T) * mask_h) @ v_h     (per head)
// qr/kr are bf16 (N x C / slen x C); vt is bf16 transposed [C][slen].
// V tile staged into LDS via async global->LDS DMA when available.
// ---------------------------------------------------------------------------
template <int D>
__global__ void __launch_bounds__(256) retention_kernel(
    const __bf16* __restrict__ qr, const __bf16* __restrict__ kr,
    const __bf16* __restrict__ vt, const float* __restrict__ maskb,
    float* __restrict__ o, int N, int slen, int C, int ratio, int halfR,
    int halfN) {
  extern __shared__ __bf16 smem[];
  const int SLP   = slen + 8;                // +16B pad: async chunks stay aligned
  __bf16* vlds    = smem;                    // [D][slen+8]
  __bf16* retslab = smem + (size_t)D * SLP;  // 8 waves * 16 * 40

  const int tid = threadIdx.x;
  const int h   = blockIdx.y;

  // stage V tile: vlds[j][s] = vt[h*D + j][s]
#if HAVE_ASYNC_LDS
  {
    const int spc = slen / 8;  // 16B chunks per row
    const int chunks = D * spc;
    for (int e = tid; e < chunks; e += 256) {
      int j = e / spc, sc = (e % spc) * 8;
      const __bf16* src = vt + (size_t)(h * D + j) * slen + sc;
      __bf16* dst = vlds + (size_t)j * SLP + sc;
      __builtin_amdgcn_global_load_async_to_lds_b128(
          (GLOBAL_AS v4i*)((GLOBAL_AS void*)const_cast<__bf16*>(src)),
          (LDS_AS v4i*)((LDS_AS void*)dst), 0, 0);
    }
#if __has_builtin(__builtin_amdgcn_s_wait_asynccnt)
    __builtin_amdgcn_s_wait_asynccnt(0);
#else
    asm volatile("s_wait_asynccnt 0x0" ::: "memory");
#endif
  }
#else
  for (int e = tid; e < D * slen; e += 256) {
    int j = e / slen, s = e % slen;
    vlds[(size_t)j * SLP + s] = vt[(size_t)(h * D + j) * slen + s];
  }
#endif
  __syncthreads();

  const int wave = tid >> 5, lane = tid & 31;
  const int l16  = lane & 15;
  const int kh   = (lane & 16) ? 8 : 0;
  const int tile = blockIdx.x * 8 + wave;
  const int row0 = tile * 16;
  __bf16* myslab = retslab + wave * 16 * 40;

  // qr A-fragments (row = l16 of this tile)
  v16bf qa[D / 32];
  const __bf16* qrow = qr + (size_t)(row0 + l16) * C + h * D;
#pragma unroll
  for (int kk = 0; kk < D / 32; ++kk) qa[kk] = load_frag_bf(qrow + kk * 32, kh);

  // expanded-row -> base-row map for this lane's 8 C-frag rows
  int brow[8];
#pragma unroll
  for (int r = 0; r < 8; ++r) {
    int n  = row0 + r + ((lane & 16) ? 8 : 0);
    int rr = (n < halfN) ? (n % halfR) : (halfR + (n - halfN) % halfR);
    brow[r] = rr / ratio;
  }

  v8f acc[D / 16];
#pragma unroll
  for (int i = 0; i < D / 16; ++i) acc[i] = v8f_zero();

  for (int s0 = 0; s0 < slen; s0 += 32) {
#pragma unroll
    for (int sub = 0; sub < 2; ++sub) {
      const int scol     = s0 + sub * 16 + l16;
      const __bf16* krow = kr + (size_t)scol * C + h * D;
      v8f rc = v8f_zero();
#pragma unroll
      for (int kk = 0; kk < D / 32; ++kk) {
        v16bf kb = load_frag_bf(krow + kk * 32, kh);
        rc = wmma_bf16(qa[kk], kb, rc);
      }
      // decay-mask multiply + relayout C-frag -> bf16 slab (A layout source)
#pragma unroll
      for (int r = 0; r < 8; ++r) {
        float m = maskb[((size_t)h * slen + brow[r]) * slen + scol];
        int rr  = r + ((lane & 16) ? 8 : 0);
        myslab[rr * 40 + sub * 16 + l16] = (__bf16)(rc[r] * m);
      }
    }
    // same-wave LDS ops are in-order: read slab back as A fragment
    v16bf a2 = load_frag_bf(myslab + l16 * 40, kh);
#pragma unroll
    for (int dt = 0; dt < D / 16; ++dt) {
      const __bf16* vcol = vlds + (size_t)(dt * 16 + l16) * SLP + s0;
      v16bf b2 = load_frag_bf(vcol, kh);
      acc[dt] = wmma_bf16(a2, b2, acc[dt]);
    }
  }

#pragma unroll
  for (int dt = 0; dt < D / 16; ++dt)
#pragma unroll
    for (int r = 0; r < 8; ++r) {
      int n = row0 + r + ((lane & 16) ? 8 : 0);
      o[(size_t)n * C + h * D + dt * 16 + l16] = acc[dt][r];
    }
}

// ---------------------------------------------------------------------------
// norms / elementwise / conv / tables
// ---------------------------------------------------------------------------
__global__ void __launch_bounds__(256) layernorm_kernel(
    const float* __restrict__ x, const float* __restrict__ w,
    const float* __restrict__ b, float* __restrict__ out, int C) {
  const int wave = threadIdx.x >> 5, lane = threadIdx.x & 31;
  const size_t row = (size_t)blockIdx.x * 8 + wave;
  const float* xr = x + row * C;
  float s = 0.f;
  for (int j = lane; j < C; j += 32) s += xr[j];
  float mean = wave_sum(s) / C;
  float vv = 0.f;
  for (int j = lane; j < C; j += 32) { float d = xr[j] - mean; vv += d * d; }
  float inv = rsqrtf(wave_sum(vv) / C + 1e-5f);
  float* orow = out + row * C;
  for (int j = lane; j < C; j += 32) orow[j] = (xr[j] - mean) * inv * w[j] + b[j];
}

__global__ void __launch_bounds__(256) groupnorm_kernel(
    const float* __restrict__ x, const float* __restrict__ w,
    const float* __restrict__ b, float* __restrict__ out, int C, int heads) {
  const int wave = threadIdx.x >> 5, lane = threadIdx.x & 31;
  const size_t rh = (size_t)blockIdx.x * 8 + wave;
  const int d = C / heads;
  const size_t n = rh / heads;
  const int h = (int)(rh % heads);
  const float* xr = x + n * C + h * d;
  float s = 0.f;
  for (int j = lane; j < d; j += 32) s += xr[j];
  float mean = wave_sum(s) / d;
  float vv = 0.f;
  for (int j = lane; j < d; j += 32) { float q = xr[j] - mean; vv += q * q; }
  float inv = rsqrtf(wave_sum(vv) / d + 1e-5f);
  float* orow = out + n * C + h * d;
  for (int j = lane; j < d; j += 32)
    orow[j] = (xr[j] - mean) * inv * w[h * d + j] + b[h * d + j];
}

__global__ void __launch_bounds__(256) pool_kernel(
    const float* __restrict__ u, float* __restrict__ xp, int W, int Wp, int C,
    int ratio, float scale) {
  int i = blockIdx.x * 256 + threadIdx.x;
  int sp = i / C, c = i % C;
  int ph = sp / Wp, pw = sp % Wp;
  float s = 0.f;
  for (int a = 0; a < ratio; ++a)
    for (int b = 0; b < ratio; ++b)
      s += u[((size_t)(ph * ratio + a) * W + pw * ratio + b) * C + c];
  xp[i] = s * scale;
}

// theta-shift (rotary) producing bf16 output for the WMMA consumers
__global__ void __launch_bounds__(256) theta_shift_kernel(
    const float* __restrict__ x, const float* __restrict__ sinT,
    const float* __restrict__ cosT, __bf16* __restrict__ out, int rows, int C,
    int d, int ratio, int halfR, int halfN, int mapped) {
  int i = blockIdx.x * 256 + threadIdx.x;
  if (i >= rows * (C / 2)) return;
  int n = i / (C / 2), p = i % (C / 2);
  int col = 2 * p, j = col % d;
  int br = n;
  if (mapped) {
    int rr = (n < halfN) ? (n % halfR) : (halfR + (n - halfN) % halfR);
    br = rr / ratio;
  }
  float s0 = sinT[(size_t)br * d + j], s1 = sinT[(size_t)br * d + j + 1];
  float c0 = cosT[(size_t)br * d + j], c1 = cosT[(size_t)br * d + j + 1];
  float x0 = x[(size_t)n * C + col], x1 = x[(size_t)n * C + col + 1];
  bf16x2 pr;
  pr[0] = (__bf16)(x0 * c0 - x1 * s0);
  pr[1] = (__bf16)(x1 * c1 + x0 * s1);
  *(bf16x2*)(out + (size_t)n * C + col) = pr;
}

// V: [slen][C] f32 -> transposed bf16 [C][slen]
__global__ void __launch_bounds__(256) convert_vt_kernel(
    const float* __restrict__ v, __bf16* __restrict__ vt, int slen, int C) {
  int i = blockIdx.x * 256 + threadIdx.x;
  if (i >= C * slen) return;
  int c = i / slen, s = i % slen;
  vt[i] = (__bf16)v[(size_t)s * C + c];
}

__global__ void __launch_bounds__(256) sincos_kernel(float* __restrict__ sinT,
                                                     float* __restrict__ cosT,
                                                     int slen, int d) {
  int i = blockIdx.x * 256 + threadIdx.x;
  if (i >= slen * d) return;
  int s = i / d, j = i % d;
  int half = d / 2;
  float lin = (half > 1) ? ((float)(j >> 1) / (float)(half - 1)) : 0.f;
  float ang = powf(10000.f, -lin);
  float t = (float)s * ang;
  sinT[i] = sinf(t);
  cosT[i] = cosf(t);
}

__global__ void __launch_bounds__(256) maskbase_kernel(float* __restrict__ maskb,
                                                       int slen, int Wimg) {
  __shared__ float red[8];
  const int r = blockIdx.x, h = blockIdx.y;
  const int tid = threadIdx.x, wave = tid >> 5, lane = tid & 31;
  const float decay = logf(1.f - exp2f(-5.f - (float)h));
  const int rr = r / Wimg, rc = r % Wimg;
  float part = 0.f;
  for (int s = tid; s < slen; s += 256) {
    float dist = fabsf((float)(rr - s / Wimg)) + fabsf((float)(rc - s % Wimg));
    part += expf(dist * decay);
  }
  part = wave_sum(part);
  if (lane == 0) red[wave] = part;
  __syncthreads();
  if (wave == 0) {
    float v = (lane < 8) ? red[lane] : 0.f;
    v = wave_sum(v);
    if (lane == 0) red[0] = v;
  }
  __syncthreads();
  const float rn = rsqrtf(red[0]);
  float* row = maskb + ((size_t)h * slen + r) * slen;
  for (int s = tid; s < slen; s += 256) {
    float dist = fabsf((float)(rr - s / Wimg)) + fabsf((float)(rc - s % Wimg));
    row[s] = expf(dist * decay) * rn;
  }
}

__global__ void __launch_bounds__(128) conv3x3_kernel(
    const float* __restrict__ xin /*NHWC*/, const float* __restrict__ wgt /*OIHW*/,
    const float* __restrict__ bias, float* __restrict__ y /*tokens x Cout*/,
    int Hin, int Win, int Cin, int Wout, int Cout, int stride) {
  extern __shared__ float patch[];  // [Cin][3][3], matches OIHW inner order
  const int tok = blockIdx.x;
  const int oh = tok / Wout, ow = tok % Wout;
  const int co = blockIdx.y * 128 + threadIdx.x;
  const int R = Cin * 9;
  for (int idx = threadIdx.x; idx < R; idx += 128) {
    int ci = idx / 9, kk = idx % 9, khh = kk / 3, kw = kk % 3;
    int ih = oh * stride - 1 + khh, iw = ow * stride - 1 + kw;
    float v = 0.f;
    if (ih >= 0 && ih < Hin && iw >= 0 && iw < Win)
      v = xin[((size_t)ih * Win + iw) * Cin + ci];
    patch[idx] = v;
  }
  __syncthreads();
  const float* wrow = wgt + (size_t)co * R;
  float acc = bias[co];
  for (int r = 0; r < R; ++r) acc += patch[r] * wrow[r];
  y[(size_t)tok * Cout + co] = acc;
}

__global__ void __launch_bounds__(256) transpose_chw_hwc_kernel(
    const float* __restrict__ in, float* __restrict__ out, int C, int HW) {
  int i = blockIdx.x * 256 + threadIdx.x;  // i = hw*C + c (coalesced writes)
  if (i >= C * HW) return;
  int c = i % C, hw = i / C;
  out[i] = in[(size_t)c * HW + hw];
}

__global__ void __launch_bounds__(256) gate_mul_kernel(float* __restrict__ o,
                                                       const float* __restrict__ g,
                                                       int n) {
  int i = blockIdx.x * 256 + threadIdx.x;
  if (i < n) o[i] = o[i] * g[i];
}

// ---------------------------------------------------------------------------
// host orchestration
// ---------------------------------------------------------------------------
static inline void launch_gemm(int epi, const float* A, const float* Wt,
                               const float* bias, const float* res, float* out,
                               int M, int Ncol, int K, float postScale,
                               hipStream_t stream) {
  dim3 grid(Ncol / 64, M / 128), block(256);
  switch (epi) {
    case 0: gemm_kernel<0><<<grid, block, 0, stream>>>(A, Wt, bias, res, out, M, Ncol, K, postScale); break;
    case 1: gemm_kernel<1><<<grid, block, 0, stream>>>(A, Wt, bias, res, out, M, Ncol, K, postScale); break;
    case 2: gemm_kernel<2><<<grid, block, 0, stream>>>(A, Wt, bias, res, out, M, Ncol, K, postScale); break;
    case 3: gemm_kernel<3><<<grid, block, 0, stream>>>(A, Wt, bias, res, out, M, Ncol, K, postScale); break;
    default: gemm_kernel<4><<<grid, block, 0, stream>>>(A, Wt, bias, res, out, M, Ncol, K, postScale); break;
  }
}

extern "C" void kernel_launch(void* const* d_in, const int* in_sizes, int n_in,
                              void* d_out, int out_size, void* d_ws, size_t ws_size,
                              hipStream_t stream) {
  (void)in_sizes; (void)n_in; (void)out_size; (void)ws_size;

  const int EMBED[4] = {128, 128, 256, 512};
  const int HEADS[4] = {1, 2, 4, 8};
  const int MLP[4]   = {256, 256, 512, 1024};
  const int NB[4]    = {3, 4, 6, 3};
  const int RATIO[4] = {16, 8, 2, 1};
  const int STR[4]   = {1, 2, 2, 2};

  // ---- workspace carve (floats); peak ~370 MB ----
  float* ws = (float*)d_ws;
  size_t pos = 0;
  auto alloc = [&](size_t n) { float* p = ws + pos; pos += n; return p; };
  const size_t NC = 8388608, NH = 16777216, SC = 524288, SD = 131072, MB = 8388608;
  float* xnhwc = alloc(NC);
  float* ybuf  = alloc(NC);
  float* tbuf  = alloc(NC);
  float* ubuf  = alloc(NC);
  float* qbuf  = alloc(NC);
  float* gbuf  = alloc(NC);
  float* obuf  = alloc(NC);
  float* hbuf  = alloc(NH);
  float* xpbuf = alloc(SC);
  float* kbuf  = alloc(SC);
  float* vbuf  = alloc(SC);
  float* sinT  = alloc(SD);
  float* cosT  = alloc(SD);
  float* maskb = alloc(MB);
  __bf16* qrbf = (__bf16*)alloc(NC / 2 + 16);  // N x C bf16
  __bf16* krbf = (__bf16*)alloc(SC / 2 + 16);  // slen x C bf16
  __bf16* vtbf = (__bf16*)alloc(SC / 2 + 16);  // [C][slen] bf16

  // large dynamic LDS for the fused retention kernel
  (void)hipFuncSetAttribute(reinterpret_cast<const void*>(&retention_kernel<64>),
                            hipFuncAttributeMaxDynamicSharedMemorySize, 160 * 1024);
  (void)hipFuncSetAttribute(reinterpret_cast<const void*>(&retention_kernel<128>),
                            hipFuncAttributeMaxDynamicSharedMemorySize, 160 * 1024);

  auto P = [&](int i) { return (const float*)d_in[i]; };
  int pidx = 1;  // d_in[0] = x

  // stage-1 input: NCHW (1,128,64,1024) -> NHWC tokens
  {
    int total = 128 * 64 * 1024;
    transpose_chw_hwc_kernel<<<(total + 255) / 256, 256, 0, stream>>>(
        (const float*)d_in[0], xnhwc, 128, 64 * 1024);
  }

  float* outf = (float*)d_out;
  size_t out_off = 0;
  const float* stage_in = xnhwc;
  int H = 64, W = 1024, Cin = 128;

  for (int s = 0; s < 4; ++s) {
    const int C = EMBED[s], heads = HEADS[s], ratio = RATIO[s], str = STR[s];
    const int Hout = (H + 2 - 3) / str + 1;
    const int Wout = (W + 2 - 3) / str + 1;
    const int N = Hout * Wout;
    const int d = C / heads;
    const int Wp = Wout / ratio, Hp = Hout / ratio;
    const int slen = Hp * Wp;
    const int halfR = slen * ratio / 2, halfN = N / 2;

    const float* cw  = P(pidx++); const float* cb  = P(pidx++);
    const float* pew = P(pidx++); const float* peb = P(pidx++);
    const float* nw  = P(pidx++); const float* nb  = P(pidx++);

    // conv stem -> tokens, then patch-embed LN
    conv3x3_kernel<<<dim3(N, C / 128), 128, Cin * 9 * sizeof(float), stream>>>(
        stage_in, cw, cb, ybuf, H, W, Cin, Wout, C, str);
    layernorm_kernel<<<N / 8, 256, 0, stream>>>(ybuf, pew, peb, tbuf, C);

    // relative-position tables for this stage
    sincos_kernel<<<(slen * d + 255) / 256, 256, 0, stream>>>(sinT, cosT, slen, d);
    maskbase_kernel<<<dim3(slen, heads), 256, 0, stream>>>(maskb, slen, Wout);

    const float kscale = rsqrtf((float)d);
    const size_t shmem =
        (size_t)d * (slen + 8) * sizeof(__bf16) + 8 * 16 * 40 * sizeof(__bf16);

    for (int blk = 0; blk < NB[s]; ++blk) {
      const float* n1w = P(pidx++); const float* n1b = P(pidx++);
      const float* n2w = P(pidx++); const float* n2b = P(pidx++);
      const float* wq = P(pidx++); const float* bq = P(pidx++);
      const float* wk = P(pidx++); const float* bk = P(pidx++);
      const float* wv = P(pidx++); const float* bv = P(pidx++);
      const float* wg = P(pidx++); const float* bg = P(pidx++);
      const float* wo = P(pidx++); const float* bo = P(pidx++);
      const float* gnw = P(pidx++); const float* gnb = P(pidx++);
      const float* w1 = P(pidx++); const float* b1 = P(pidx++);
      const float* w2 = P(pidx++); const float* b2 = P(pidx++);

      // ---- MSR ----
      layernorm_kernel<<<N / 8, 256, 0, stream>>>(tbuf, n1w, n1b, ubuf, C);
      launch_gemm(0, ubuf, wq, bq, nullptr, qbuf, N, C, C, 0.f, stream);
      pool_kernel<<<(slen * C) / 256, 256, 0, stream>>>(
          ubuf, xpbuf, Wout, Wp, C, ratio, 1.f / (float)(ratio * ratio));
      launch_gemm(1, xpbuf, wk, bk, nullptr, kbuf, slen, C, C, kscale, stream);
      launch_gemm(0, xpbuf, wv, bv, nullptr, vbuf, slen, C, C, 0.f, stream);
      theta_shift_kernel<<<(N * (C / 2) + 255) / 256, 256, 0, stream>>>(
          qbuf, sinT, cosT, qrbf, N, C, d, ratio, halfR, halfN, 1);
      theta_shift_kernel<<<(slen * (C / 2) + 255) / 256, 256, 0, stream>>>(
          kbuf, sinT, cosT, krbf, slen, C, d, 1, 0, 0, 0);
      convert_vt_kernel<<<(C * slen) / 256, 256, 0, stream>>>(vbuf, vtbf, slen, C);
      if (d == 128)
        retention_kernel<128><<<dim3(N / 128, heads), 256, shmem, stream>>>(
            qrbf, krbf, vtbf, maskb, obuf, N, slen, C, ratio, halfR, halfN);
      else
        retention_kernel<64><<<dim3(N / 128, heads), 256, shmem, stream>>>(
            qrbf, krbf, vtbf, maskb, obuf, N, slen, C, ratio, halfR, halfN);
      groupnorm_kernel<<<((size_t)N * heads) / 8, 256, 0, stream>>>(
          obuf, gnw, gnb, obuf, C, heads);
      launch_gemm(2, ubuf, wg, bg, nullptr, gbuf, N, C, C, 0.f, stream);  // silu gate
      gate_mul_kernel<<<((size_t)N * C) / 256, 256, 0, stream>>>(obuf, gbuf, N * C);
      launch_gemm(4, obuf, wo, bo, tbuf, tbuf, N, C, C, 0.f, stream);  // +residual

      // ---- MLP ----
      layernorm_kernel<<<N / 8, 256, 0, stream>>>(tbuf, n2w, n2b, ubuf, C);
      launch_gemm(3, ubuf, w1, b1, nullptr, hbuf, N, MLP[s], C, 0.f, stream);
      launch_gemm(4, hbuf, w2, b2, tbuf, tbuf, N, C, MLP[s], 0.f, stream);
    }

    // stage output (final LN) directly into d_out; next stage reads it as NHWC
    float* stage_out = outf + out_off;
    layernorm_kernel<<<N / 8, 256, 0, stream>>>(tbuf, nw, nb, stage_out, C);
    stage_in = stage_out;
    out_off += (size_t)N * C;
    H = Hout; W = Wout; Cin = C;
  }
}
